// CopynetDecoderRNN_1082331758647
// MI455X (gfx1250) — matrix operand order repaired and checked by reading, not compiled, not run on hardware
//
#include <hip/hip_runtime.h>

// Problem constants
#define Bn 64
#define Ln 1024
#define En 300
#define Hn 256
#define DEC_Vn 30000
#define Vn 50000
#define INn (2*Hn+En)   // 812

typedef __attribute__((ext_vector_type(16))) __bf16 v16bf;
typedef __attribute__((ext_vector_type(8)))  float  v8f;

#define NEG_INF (-__builtin_inff())

__device__ __forceinline__ float sigf(float x) { return 1.0f / (1.0f + __expf(-x)); }

// Build A-fragment (16-bit A 16x32 layout): lane<16 -> k chunks [kb..kb+7],[kb+16..kb+23] with kb=ks*32;
// lane>=16 -> kb=ks*32+8. Source is a contiguous fp32 row.
__device__ __forceinline__ v16bf make_a_frag(const float* row, int ks, int lane) {
    int kb = ks * 32 + ((lane < 16) ? 0 : 8);
    v16bf v;
#pragma unroll
    for (int i = 0; i < 8; ++i) v[i] = (__bf16)row[kb + i];
#pragma unroll
    for (int i = 0; i < 8; ++i) v[8 + i] = (__bf16)row[kb + 16 + i];
    return v;
}

// Build B-fragment (16-bit B 32x16 layout): lane holds column n = lane%16,
// k chunks [kb..kb+7],[kb+16..kb+23] with kb = ks*32 + (lane<16 ? 0 : 8). Source row = weight row n (row-major W[n][k]).
__device__ __forceinline__ v16bf make_b_frag(const float* wrow, int ks, int lane) {
    int kb = ks * 32 + ((lane < 16) ? 0 : 8);
    v16bf v;
#pragma unroll
    for (int i = 0; i < 8; ++i) v[i] = (__bf16)wrow[kb + i];
#pragma unroll
    for (int i = 0; i < 8; ++i) v[8 + i] = (__bf16)wrow[kb + 16 + i];
    return v;
}

// ---------------- K1: selective read  sel[b,h] = mean over matching l of enc[b,l,h] ----------------
__global__ void k_sel(const int* __restrict__ input_id, const int* __restrict__ enc_ids,
                      const float* __restrict__ enc, float* __restrict__ sel) {
    int b = blockIdx.x, h = threadIdx.x;
    int id = input_id[b];
    int cnt = 0; float acc = 0.f;
    for (int l = 0; l < Ln; ++l) {
        if (enc_ids[b * Ln + l] == id) { cnt++; acc += enc[((size_t)b * Ln + l) * Hn + h]; }
    }
    sel[b * Hn + h] = (cnt > 0) ? acc / (float)cnt : 0.f;
}

// ---------------- K2: LSTM cell 0 ----------------
__global__ void k_lstm0(const float* __restrict__ x_embed, const float* __restrict__ attention,
                        const float* __restrict__ sel, const float* __restrict__ h0,
                        const float* __restrict__ c0,
                        const float* __restrict__ w_ih, const float* __restrict__ w_hh,
                        const float* __restrict__ b_ih, const float* __restrict__ b_hh,
                        float* __restrict__ h1, float* __restrict__ c1,
                        float* __restrict__ out_h, float* __restrict__ out_c) {
    __shared__ float xs[INn];
    __shared__ float hs[Hn];
    __shared__ float g[4 * Hn];
    int b = blockIdx.x, t = threadIdx.x;
    for (int i = t; i < En; i += 256) xs[i] = x_embed[b * En + i];
    for (int i = t; i < Hn; i += 256) {
        xs[En + i]      = attention[b * Hn + i];
        xs[En + Hn + i] = sel[b * Hn + i];
        hs[i]           = h0[b * Hn + i];          // layer 0 slice passed in
    }
    __syncthreads();
#pragma unroll
    for (int j = 0; j < 4; ++j) {
        int f = j * 256 + t;
        const float* wr = w_ih + (size_t)f * INn;
        const float* hr = w_hh + (size_t)f * Hn;
        float a = b_ih[f] + b_hh[f];
        for (int k = 0; k < INn; ++k) a += wr[k] * xs[k];
        for (int k = 0; k < Hn; ++k)  a += hr[k] * hs[k];
        g[f] = a;
    }
    __syncthreads();
    {
        float gi = g[t], gf = g[Hn + t], gg = g[2 * Hn + t], go = g[3 * Hn + t];
        float cold = c0[b * Hn + t];
        float cn = sigf(gf) * cold + sigf(gi) * tanhf(gg);
        float hn = sigf(go) * tanhf(cn);
        h1[b * Hn + t] = hn; c1[b * Hn + t] = cn;
        out_h[b * Hn + t] = hn; out_c[b * Hn + t] = cn;
    }
}

// ---------------- K3: LSTM cell 1 + attention projection q = h2 @ attn_w^T + attn_b ----------------
__global__ void k_lstm1(const float* __restrict__ x, const float* __restrict__ h0l,
                        const float* __restrict__ c0l,
                        const float* __restrict__ w_ih, const float* __restrict__ w_hh,
                        const float* __restrict__ b_ih, const float* __restrict__ b_hh,
                        const float* __restrict__ attn_w, const float* __restrict__ attn_b,
                        float* __restrict__ h2, float* __restrict__ c2,
                        float* __restrict__ out_h, float* __restrict__ out_c,
                        float* __restrict__ qv) {
    __shared__ float xs[Hn];
    __shared__ float hs[Hn];
    __shared__ float g[4 * Hn];
    __shared__ float h2s[Hn];
    int b = blockIdx.x, t = threadIdx.x;
    xs[t] = x[b * Hn + t];
    hs[t] = h0l[b * Hn + t];
    __syncthreads();
#pragma unroll
    for (int j = 0; j < 4; ++j) {
        int f = j * 256 + t;
        const float* wr = w_ih + (size_t)f * Hn;
        const float* hr = w_hh + (size_t)f * Hn;
        float a = b_ih[f] + b_hh[f];
        for (int k = 0; k < Hn; ++k) a += wr[k] * xs[k] + hr[k] * hs[k];
        g[f] = a;
    }
    __syncthreads();
    {
        float gi = g[t], gf = g[Hn + t], gg = g[2 * Hn + t], go = g[3 * Hn + t];
        float cold = c0l[b * Hn + t];
        float cn = sigf(gf) * cold + sigf(gi) * tanhf(gg);
        float hn = sigf(go) * tanhf(cn);
        h2[b * Hn + t] = hn; c2[b * Hn + t] = cn;
        out_h[b * Hn + t] = hn; out_c[b * Hn + t] = cn;
        h2s[t] = hn;
    }
    __syncthreads();
    {
        const float* wr = attn_w + (size_t)t * Hn;
        float a = attn_b[t];
        for (int k = 0; k < Hn; ++k) a += wr[k] * h2s[k];
        qv[b * Hn + t] = a;
    }
}

// ---------------- K4: attention scores + softmax + applied ----------------
__global__ void k_attn(const float* __restrict__ qv, const float* __restrict__ enc,
                       const int* __restrict__ enc_ids, float* __restrict__ applied) {
    __shared__ float qs[Hn];
    __shared__ float aw[Ln];
    __shared__ float red[256];
    int b = blockIdx.x, t = threadIdx.x;
    qs[t] = qv[b * Hn + t];
    __syncthreads();
    float sc[4];
#pragma unroll
    for (int j = 0; j < 4; ++j) {
        int l = j * 256 + t;
        const float* er = enc + ((size_t)b * Ln + l) * Hn;
        float a = 0.f;
        for (int k = 0; k < Hn; ++k) a += qs[k] * er[k];
        if (enc_ids[b * Ln + l] == 0) a = NEG_INF;
        sc[j] = a;
    }
    float m = fmaxf(fmaxf(sc[0], sc[1]), fmaxf(sc[2], sc[3]));
    red[t] = m; __syncthreads();
    for (int s = 128; s > 0; s >>= 1) { if (t < s) red[t] = fmaxf(red[t], red[t + s]); __syncthreads(); }
    float M = red[0]; __syncthreads();
    float ssum = 0.f;
#pragma unroll
    for (int j = 0; j < 4; ++j) {
        float e = (sc[j] == NEG_INF) ? 0.f : __expf(sc[j] - M);
        aw[j * 256 + t] = e;
        ssum += e;
    }
    red[t] = ssum; __syncthreads();
    for (int s = 128; s > 0; s >>= 1) { if (t < s) red[t] += red[t + s]; __syncthreads(); }
    float invZ = 1.f / red[0]; __syncthreads();
#pragma unroll
    for (int j = 0; j < 4; ++j) aw[j * 256 + t] *= invZ;
    __syncthreads();
    float acc = 0.f;
    for (int l = 0; l < Ln; ++l) acc += aw[l] * enc[((size_t)b * Ln + l) * Hn + t];
    applied[b * Hn + t] = acc;
}

// ---------------- K5: cur_att = tanh([applied, h2] @ comb_w^T + comb_b) ----------------
__global__ void k_comb(const float* __restrict__ applied, const float* __restrict__ h2,
                       const float* __restrict__ comb_w, const float* __restrict__ comb_b,
                       float* __restrict__ catt, float* __restrict__ out_att) {
    __shared__ float xs[2 * Hn];
    int b = blockIdx.x, t = threadIdx.x;
    xs[t] = applied[b * Hn + t];
    xs[Hn + t] = h2[b * Hn + t];
    __syncthreads();
    const float* wr = comb_w + (size_t)t * 2 * Hn;
    float a = comb_b[t];
    for (int k = 0; k < 2 * Hn; ++k) a += wr[k] * xs[k];
    a = tanhf(a);
    catt[b * Hn + t] = a;
    out_att[b * Hn + t] = a;
}

// ---------------- K6: pre-pack copy_w into bf16 B-fragments [nt(16)][ks(8)][lane(32)] ----------------
__global__ void k_prep_copyw(const float* __restrict__ copy_w, v16bf* __restrict__ frag) {
    int t = blockIdx.x * 256 + threadIdx.x;     // 4096 threads
    int lane = t & 31;
    int ks = (t >> 5) & 7;
    int nt = t >> 8;
    int n = nt * 16 + (lane & 15);
    v16bf v = make_b_frag(copy_w + (size_t)n * Hn, ks, lane);
    frag[((nt * 8 + ks) * 32) + lane] = v;
}

// ---------------- K7: gen = cur_att @ gen_w^T + gen_b  (bf16 WMMA) ----------------
__global__ void __launch_bounds__(256) k_gen(const float* __restrict__ catt,
                                             const float* __restrict__ gen_w,
                                             const float* __restrict__ gen_b,
                                             float* __restrict__ gen) {
    int wid  = (blockIdx.x * 256 + threadIdx.x) >> 5;
    int lane = threadIdx.x & 31;
    if (wid >= 1500) return;                    // wave-uniform guard (4 mtiles x 375 groups)
    int mtile = wid & 3;
    int g0 = (wid >> 2) * 5;                    // 5 ntiles per wave, 375*5 = 1875
    int mrow = mtile * 16 + (lane & 15);
    const float* arow = catt + (size_t)mrow * Hn;
    v16bf a[8];
#pragma unroll
    for (int ks = 0; ks < 8; ++ks) a[ks] = make_a_frag(arow, ks, lane);
    int mbase = mtile * 16 + ((lane < 16) ? 0 : 8);
    for (int gi = 0; gi < 5; ++gi) {
        int nt = g0 + gi;
        int n = nt * 16 + (lane & 15);
        const float* brow = gen_w + (size_t)n * Hn;
        v8f c = {};
#pragma unroll
        for (int ks = 0; ks < 8; ++ks) {
            v16bf bv = make_b_frag(brow, ks, lane);
            c = __builtin_amdgcn_wmma_f32_16x16x32_bf16(false, a[ks], false, bv,
                                                        (short)0, c, false, false);
        }
        float bias = gen_b[n];
#pragma unroll
        for (int r = 0; r < 8; ++r)
            gen[(size_t)(mbase + r) * DEC_Vn + n] = c[r] + bias;
    }
}

// ---------------- K8: cs[b,l] = sum_h sigmoid(enc @ copy_w^T + copy_b) * cur_att  (bf16 WMMA, fused) ----
__global__ void __launch_bounds__(256) k_copyscore(const float* __restrict__ enc,
                                                   const v16bf* __restrict__ cwfrag,
                                                   const float* __restrict__ copy_b,
                                                   const float* __restrict__ catt,
                                                   float* __restrict__ cs) {
    extern __shared__ char smem[];              // 131072 B: all copy_w fragments
    v16bf* sfrag = (v16bf*)smem;
    int t = threadIdx.x;
    for (int i = t; i < 4096; i += 256) sfrag[i] = cwfrag[i];
    __syncthreads();
    int wv = t >> 5, lane = t & 31;
    int blk = blockIdx.x;                       // 512 blocks x 8 waves = 4096 M-tiles
    int b = blk >> 3;
    int ltile = (blk & 7) * 8 + wv;             // 0..63
    const float* arow = enc + ((size_t)b * Ln + ltile * 16 + (lane & 15)) * Hn;
    v16bf a[8];
#pragma unroll
    for (int ks = 0; ks < 8; ++ks) a[ks] = make_a_frag(arow, ks, lane);
    float accs[8];
#pragma unroll
    for (int r = 0; r < 8; ++r) accs[r] = 0.f;
    const float* cb = catt + (size_t)b * Hn;
    for (int nt = 0; nt < 16; ++nt) {
        v8f c = {};
#pragma unroll
        for (int ks = 0; ks < 8; ++ks) {
            v16bf bv = sfrag[(nt * 8 + ks) * 32 + lane];
            c = __builtin_amdgcn_wmma_f32_16x16x32_bf16(false, a[ks], false, bv,
                                                        (short)0, c, false, false);
        }
        int n = nt * 16 + (lane & 15);
        float bias = copy_b[n];
        float cm = cb[n];
#pragma unroll
        for (int r = 0; r < 8; ++r)
            accs[r] += sigf(c[r] + bias) * cm;
    }
    // reduce over N (16 lanes per half-wave hold cols 0..15 of rows r / r+8)
#pragma unroll
    for (int r = 0; r < 8; ++r) {
        float v = accs[r];
        v += __shfl_xor(v, 1, 32);
        v += __shfl_xor(v, 2, 32);
        v += __shfl_xor(v, 4, 32);
        v += __shfl_xor(v, 8, 32);
        accs[r] = v;
    }
    if ((lane & 15) == 0) {
        int rbase = (lane >> 4) * 8;
#pragma unroll
        for (int r = 0; r < 8; ++r)
            cs[(size_t)b * Ln + ltile * 16 + rbase + r] = accs[r];
    }
}

// ---------------- K9a: joint softmax stats + gen probabilities ----------------
__global__ void k_probs(const float* __restrict__ gen, const float* __restrict__ cs,
                        const int* __restrict__ enc_ids, float* __restrict__ out,
                        float* __restrict__ mz) {
    __shared__ float red[256];
    int b = blockIdx.x, t = threadIdx.x;
    const float* gb = gen + (size_t)b * DEC_Vn;
    const float* cb = cs + (size_t)b * Ln;
    const int* ib = enc_ids + (size_t)b * Ln;
    float m = NEG_INF;
    for (int i = t; i < DEC_Vn; i += 256) m = fmaxf(m, gb[i]);
    for (int i = t; i < Ln; i += 256) if (ib[i] != 0) m = fmaxf(m, cb[i]);
    red[t] = m; __syncthreads();
    for (int s = 128; s > 0; s >>= 1) { if (t < s) red[t] = fmaxf(red[t], red[t + s]); __syncthreads(); }
    float M = red[0]; __syncthreads();
    float ssum = 0.f;
    for (int i = t; i < DEC_Vn; i += 256) ssum += __expf(gb[i] - M);
    for (int i = t; i < Ln; i += 256) if (ib[i] != 0) ssum += __expf(cb[i] - M);
    red[t] = ssum; __syncthreads();
    for (int s = 128; s > 0; s >>= 1) { if (t < s) red[t] += red[t + s]; __syncthreads(); }
    float Z = red[0];
    float invZ = 1.f / Z;
    if (t == 0) { mz[2 * b] = M; mz[2 * b + 1] = invZ; }
    float* ob = out + (size_t)b * Vn;
    for (int i = t; i < DEC_Vn; i += 256) ob[i] = __expf(gb[i] - M) * invZ;
    for (int i = DEC_Vn + t; i < Vn; i += 256) ob[i] = 0.f;
}

// ---------------- K9b: scatter copy probabilities ----------------
__global__ void k_scatter(const float* __restrict__ cs, const int* __restrict__ enc_ids,
                          const float* __restrict__ mz, float* __restrict__ out) {
    int b = blockIdx.x, t = threadIdx.x;
    float M = mz[2 * b], invZ = mz[2 * b + 1];
    for (int i = t; i < Ln; i += 256) {
        int id = enc_ids[(size_t)b * Ln + i];
        if (id != 0)
            atomicAdd(&out[(size_t)b * Vn + id], __expf(cs[(size_t)b * Ln + i] - M) * invZ);
    }
}

// ---------------- K9c: log transform ----------------
__global__ void k_log(float* __restrict__ out) {
    size_t i = (size_t)blockIdx.x * 256 + threadIdx.x;
    if (i < (size_t)Bn * Vn) {
        float v = out[i];
        out[i] = (v > 0.f) ? __logf(v) : NEG_INF;
    }
}

extern "C" void kernel_launch(void* const* d_in, const int* in_sizes, int n_in,
                              void* d_out, int out_size, void* d_ws, size_t ws_size,
                              hipStream_t stream) {
    (void)in_sizes; (void)n_in; (void)out_size; (void)ws_size;
    const int*   input_id  = (const int*)d_in[0];
    const float* x_embed   = (const float*)d_in[1];
    const float* enc       = (const float*)d_in[2];
    const int*   enc_ids   = (const int*)d_in[3];
    const float* h0        = (const float*)d_in[4];
    const float* c0        = (const float*)d_in[5];
    const float* attention = (const float*)d_in[6];
    const float* w_ih0     = (const float*)d_in[7];
    const float* w_hh0     = (const float*)d_in[8];
    const float* b_ih0     = (const float*)d_in[9];
    const float* b_hh0     = (const float*)d_in[10];
    const float* w_ih1     = (const float*)d_in[11];
    const float* w_hh1     = (const float*)d_in[12];
    const float* b_ih1     = (const float*)d_in[13];
    const float* b_hh1     = (const float*)d_in[14];
    const float* attn_w    = (const float*)d_in[15];
    const float* attn_b    = (const float*)d_in[16];
    const float* comb_w    = (const float*)d_in[17];
    const float* comb_b    = (const float*)d_in[18];
    const float* gen_w     = (const float*)d_in[19];
    const float* gen_b     = (const float*)d_in[20];
    const float* copy_w    = (const float*)d_in[21];
    const float* copy_b    = (const float*)d_in[22];

    // Workspace layout (floats)
    float* ws      = (float*)d_ws;
    float* sel     = ws;                   // 16384
    float* h1      = sel + Bn * Hn;
    float* c1      = h1 + Bn * Hn;
    float* h2      = c1 + Bn * Hn;
    float* c2      = h2 + Bn * Hn;
    float* qv      = c2 + Bn * Hn;
    float* applied = qv + Bn * Hn;
    float* catt    = applied + Bn * Hn;
    float* gen     = catt + Bn * Hn;                   // 1,920,000
    float* cs      = gen + (size_t)Bn * DEC_Vn;        // 65,536
    v16bf* cwfrag  = (v16bf*)(cs + (size_t)Bn * Ln);   // 4096 frags, 32B-aligned
    float* mz      = (float*)(cwfrag + 4096);          // 2*B

    // Output layout: out (B*V) | h_stack (2,B,H) | c_stack (2,B,H) | cur_att (B,1,H)
    float* out     = (float*)d_out;
    float* out_h   = out + (size_t)Bn * Vn;
    float* out_c   = out_h + 2 * Bn * Hn;
    float* out_att = out_c + 2 * Bn * Hn;

    // Independent of sequential chain: pack copy_w fragments early
    k_prep_copyw<<<16, 256, 0, stream>>>(copy_w, cwfrag);

    k_sel<<<Bn, 256, 0, stream>>>(input_id, enc_ids, enc, sel);
    k_lstm0<<<Bn, 256, 0, stream>>>(x_embed, attention, sel,
                                    h0 /*layer0*/, c0 /*layer0*/,
                                    w_ih0, w_hh0, b_ih0, b_hh0,
                                    h1, c1, out_h, out_c);
    k_lstm1<<<Bn, 256, 0, stream>>>(h1, h0 + Bn * Hn, c0 + Bn * Hn,
                                    w_ih1, w_hh1, b_ih1, b_hh1,
                                    attn_w, attn_b,
                                    h2, c2, out_h + Bn * Hn, out_c + Bn * Hn, qv);
    k_attn<<<Bn, 256, 0, stream>>>(qv, enc, enc_ids, applied);
    k_comb<<<Bn, 256, 0, stream>>>(applied, h2, comb_w, comb_b, catt, out_att);

    k_gen<<<188, 256, 0, stream>>>(catt, gen_w, gen_b, gen);
    k_copyscore<<<512, 256, 131072, stream>>>(enc, cwfrag, copy_b, catt, cs);

    k_probs<<<Bn, 256, 0, stream>>>(gen, cs, enc_ids, out, mz);
    k_scatter<<<Bn, 256, 0, stream>>>(cs, enc_ids, mz, out);
    k_log<<<(Bn * Vn + 255) / 256, 256, 0, stream>>>(out);
}
